// GCN_15977278341809
// MI455X (gfx1250) — compile-verified, matrix operand order checked by reading, not run on hardware
//
#include <hip/hip_runtime.h>
#include <hip/hip_bf16.h>
#include <math.h>

// ---------------------------------------------------------------------------
// CDNA5 (gfx1250) GCN forward.
// - Node GEMMs: v_wmma_f32_16x16x32_bf16, 2x2 tiles/wave, LDS double-buffered,
//   async global->LDS staging (GLOBAL_LOAD_ASYNC_TO_LDS_B128 when available).
// - Edge scatter: fp32 L2 atomics. Segment-max: order-preserving uint atomicMax.
// ---------------------------------------------------------------------------

typedef __attribute__((ext_vector_type(16))) __bf16 v16bf;
typedef __attribute__((ext_vector_type(8)))  float  v8f;
typedef int v4i __attribute__((vector_size(16)));   // matches builtin param type

#define NN     50000
#define NE     400000
#define F_IN   128
#define DCONV  256
#define GDIM   128
#define LDIM   64
#define NGRAPH 64

#define AS1 __attribute__((address_space(1)))
#define AS3 __attribute__((address_space(3)))

#if defined(__gfx1250__) && __has_builtin(__builtin_amdgcn_global_load_async_to_lds_b128)
#define HAVE_ASYNC_LDS 1
#else
#define HAVE_ASYNC_LDS 0
#endif

static __device__ __forceinline__ unsigned short f2bf(float v) {
  // round-to-nearest-even f32 -> bf16
  unsigned b = __float_as_uint(v);
  unsigned r = (b + 0x7fffu + ((b >> 16) & 1u)) >> 16;
  return (unsigned short)r;
}

// 16-byte global->LDS copy: async DMA path when the builtin exists.
static __device__ __forceinline__ void cp16(const unsigned short* g, unsigned short* l) {
#if HAVE_ASYNC_LDS
  __builtin_amdgcn_global_load_async_to_lds_b128((AS1 v4i*)g, (AS3 v4i*)l, 0, 0);
#else
  *(uint4*)l = *(const uint4*)g;
#endif
}

static __device__ __forceinline__ void stage_join() {
#if HAVE_ASYNC_LDS
#if __has_builtin(__builtin_amdgcn_s_wait_asynccnt)
  __builtin_amdgcn_s_wait_asynccnt(0);
#else
  asm volatile("s_wait_asynccnt 0" ::: "memory");
#endif
#endif
  __syncthreads();
}

// ---------------- utility fills / converts ----------------------------------

__global__ void k_fill_f32(float* __restrict__ p, long n, float v) {
  long i = (long)blockIdx.x * blockDim.x + threadIdx.x;
  if (i < n) p[i] = v;
}

__global__ void k_fill_u32(unsigned int* __restrict__ p, long n, unsigned int v) {
  long i = (long)blockIdx.x * blockDim.x + threadIdx.x;
  if (i < n) p[i] = v;
}

__global__ void k_f32_to_bf16(const float* __restrict__ in,
                              unsigned short* __restrict__ out,
                              long n, int relu) {
  long i = (long)blockIdx.x * blockDim.x + threadIdx.x;
  if (i >= n) return;
  float v = in[i];
  if (relu) v = v > 0.f ? v : 0.f;
  out[i] = f2bf(v);
}

// W [din, dout] f32 -> wbT [dout, din] bf16 (transposed for row-copy staging)
__global__ void k_wT_bf16(const float* __restrict__ W, unsigned short* __restrict__ wbT,
                          int din, int dout) {
  long i = (long)blockIdx.x * blockDim.x + threadIdx.x;
  if (i >= (long)din * dout) return;
  int n = (int)(i / din), k = (int)(i % din);
  wbT[i] = f2bf(W[(long)k * dout + n]);
}

// ---------------- degree / normalization ------------------------------------

__global__ void k_deg(const int* __restrict__ ei, float* __restrict__ deg) {
  long e = (long)blockIdx.x * blockDim.x + threadIdx.x;
  if (e >= NE) return;
  atomicAdd(&deg[ei[NE + e]], 1.0f);   // target index = col = edge_index[1]
}

__global__ void k_rsqrt_inplace(float* __restrict__ d, long n) {
  long i = (long)blockIdx.x * blockDim.x + threadIdx.x;
  if (i >= n) return;
  float v = d[i];
  d[i] = v > 0.f ? rsqrtf(v) : 0.f;
}

__global__ void k_tile_glob(const float* __restrict__ g0, float* __restrict__ g) {
  int i = blockIdx.x * blockDim.x + threadIdx.x;   // NGRAPH*GDIM
  if (i < NGRAPH * GDIM) g[i] = g0[i % GDIM];
}

// ---------------- WMMA node GEMM --------------------------------------------
// hout[m,n] = sum_k hb[m,k]*wbT[n,k] + bias[n] + gnode[ga[m],n]
// 256 thr (8 waves as 4x2), block tile 128(M) x 64(N), wave tile 32x32
// (2x2 WMMA 16x16x32 bf16), double-buffered LDS with async staging.

__global__ __launch_bounds__(256)
void k_wmma_gemm(const unsigned short* __restrict__ hb,   // [nrows, din] bf16
                 const unsigned short* __restrict__ wbT,  // [dout, din]  bf16
                 const float* __restrict__ bias,          // [dout]
                 const float* __restrict__ gnode,         // [NGRAPH, dout]
                 const int*  __restrict__ ga,             // [nrows]
                 float* __restrict__ hout,                // [nrows, dout]
                 int nrows, int din, int dout) {
  __shared__ unsigned short As[2][128 * 32];   // [m][k]
  __shared__ unsigned short Bs[2][64 * 32];    // [n][k]

  const int tid   = threadIdx.x;
  const int lane  = tid & 31;
  const int wave  = tid >> 5;
  const int waveM = wave & 3;                  // 0..3
  const int waveN = wave >> 2;                 // 0..1
  const int nBlocks = dout >> 6;
  const int mBlock  = (int)(blockIdx.x / nBlocks) * 128;
  const int nBlock  = (int)(blockIdx.x % nBlocks) * 64;

  // staging: per thread, 2 A rows-chunks + 1 B row-chunk of 8 bf16 (16B)
  const int sM0 = tid >> 2;                    // 0..63
  const int sK  = (tid & 3) * 8;               // 0,8,16,24
  const int sM1 = sM0 + 64;
  const long gA0 = (long)min(mBlock + sM0, nrows - 1) * din + sK;
  const long gA1 = (long)min(mBlock + sM1, nrows - 1) * din + sK;
  const long gB  = (long)(nBlock + sM0) * din + sK;

  v8f acc00 = {}, acc01 = {}, acc10 = {}, acc11 = {};

  // issue one K-tile's staging copies into buffer `buf`
  auto stage = [&](int buf, int k0) {
    cp16(hb  + gA0 + k0, &As[buf][sM0 * 32 + sK]);
    cp16(hb  + gA1 + k0, &As[buf][sM1 * 32 + sK]);
    cp16(wbT + gB  + k0, &Bs[buf][sM0 * 32 + sK]);
  };

  stage(0, 0);
  stage_join();

  // Fragment addressing per ISA 7.12.2 (16-bit 16x32): lanes 0-15 -> K base 0,
  // lanes 16-31 -> K base 8; VGPR 0-3 pairs K+0..7, VGPR 4-7 pairs K+16..23.
  const int mloc = waveM * 32 + (lane & 15);
  const int nloc = waveN * 32 + (lane & 15);
  const int kgrp = (lane >> 4) * 8;

  int buf = 0;
  for (int k0 = 0; k0 < din; k0 += 32) {
    if (k0 + 32 < din) stage(buf ^ 1, k0 + 32);   // prefetch next K-tile

    union { v16bf v; unsigned int u[8]; } a0, a1, b0, b1;
#pragma unroll
    for (int v = 0; v < 8; ++v) {
      int k = ((v & 4) ? 16 : 0) + kgrp + (v & 3) * 2;
      a0.u[v] = *(const unsigned int*)(&As[buf][(mloc     ) * 32 + k]);
      a1.u[v] = *(const unsigned int*)(&As[buf][(mloc + 16) * 32 + k]);
      b0.u[v] = *(const unsigned int*)(&Bs[buf][(nloc     ) * 32 + k]);
      b1.u[v] = *(const unsigned int*)(&Bs[buf][(nloc + 16) * 32 + k]);
    }
    acc00 = __builtin_amdgcn_wmma_f32_16x16x32_bf16(false, a0.v, false, b0.v, (short)0, acc00, false, false);
    acc01 = __builtin_amdgcn_wmma_f32_16x16x32_bf16(false, a0.v, false, b1.v, (short)0, acc01, false, false);
    acc10 = __builtin_amdgcn_wmma_f32_16x16x32_bf16(false, a1.v, false, b0.v, (short)0, acc10, false, false);
    acc11 = __builtin_amdgcn_wmma_f32_16x16x32_bf16(false, a1.v, false, b1.v, (short)0, acc11, false, false);

    stage_join();
    buf ^= 1;
  }

  // Epilogue. C layout: VGPR r -> M = r (lanes 0-15) / r+8 (lanes 16-31), N = lane&15.
  const int n0 = nBlock + nloc;
  const int n1 = n0 + 16;
  const int mRow = mBlock + waveM * 32 + ((lane >> 4) * 8);
  const float bn0 = bias[n0], bn1 = bias[n1];
#pragma unroll
  for (int r = 0; r < 8; ++r) {
    int m0 = mRow + r;
    int m1 = m0 + 16;
    if (m0 < nrows) {
      long base = (long)m0 * dout;
      long gb   = (long)ga[m0] * dout;
      hout[base + n0] = acc00[r] + bn0 + gnode[gb + n0];
      hout[base + n1] = acc01[r] + bn1 + gnode[gb + n1];
    }
    if (m1 < nrows) {
      long base = (long)m1 * dout;
      long gb   = (long)ga[m1] * dout;
      hout[base + n0] = acc10[r] + bn0 + gnode[gb + n0];
      hout[base + n1] = acc11[r] + bn1 + gnode[gb + n1];
    }
  }
}

// ---------------- small global-state GEMM (M=64) ----------------------------
// C[m,n] = (acc ? C[m,n] : 0) + bias[n] + sum_k A[m,k]*B[k,n]

__global__ void k_small_gemm(const float* __restrict__ A, const float* __restrict__ B,
                             const float* __restrict__ bias, float* __restrict__ C,
                             int M, int K, int N, int acc) {
  int i = blockIdx.x * blockDim.x + threadIdx.x;
  if (i >= M * N) return;
  int m = i / N, n = i % N;
  float s = bias[n] + (acc ? C[i] : 0.f);
  const float* a = A + (long)m * K;
  for (int k = 0; k < K; ++k) s = fmaf(a[k], B[(long)k * N + n], s);
  C[i] = s;
}

// ---------------- edge scatter-add ------------------------------------------
// h_agg[col] += dinv[row]*dinv[col] * h_lin[row]; self-loops appended.

__global__ void k_scatter(const int* __restrict__ ei, const float* __restrict__ dinv,
                          const float* __restrict__ hin, float* __restrict__ hagg,
                          int dout, long total) {
  long tid = (long)blockIdx.x * blockDim.x + threadIdx.x;
  if (tid >= total) return;
  const int f4 = dout >> 2;
  long e = tid / f4;
  int  f = (int)(tid - e * f4) * 4;
  int r, c;
  if (e < NE) { r = ei[e]; c = ei[NE + e]; }
  else        { r = c = (int)(e - NE); }
  const float nm = dinv[r] * dinv[c];
  const float4 hv = *(const float4*)(hin + (long)r * dout + f);
  float* dst = hagg + (long)c * dout + f;
  atomicAdd(dst + 0, nm * hv.x);
  atomicAdd(dst + 1, nm * hv.y);
  atomicAdd(dst + 2, nm * hv.z);
  atomicAdd(dst + 3, nm * hv.w);
}

// ---------------- per-graph segment max -------------------------------------

static __device__ __forceinline__ unsigned int fenc(float f) {
  unsigned u = __float_as_uint(f);
  return (u & 0x80000000u) ? ~u : (u | 0x80000000u);
}
static __device__ __forceinline__ float fdec(unsigned int u) {
  return __uint_as_float((u & 0x80000000u) ? (u & 0x7fffffffu) : ~u);
}

__global__ void k_segmax(const float* __restrict__ h, const int* __restrict__ ga,
                         unsigned int* __restrict__ penc, int dout, long total) {
  long i = (long)blockIdx.x * blockDim.x + threadIdx.x;
  if (i >= total) return;
  long node = i / dout;
  int  f = (int)(i - node * dout);
  atomicMax(&penc[(long)ga[node] * dout + f], fenc(h[i]));
}

__global__ void k_decode(const unsigned int* __restrict__ penc,
                         float* __restrict__ parts, long n) {
  long i = (long)blockIdx.x * blockDim.x + threadIdx.x;
  if (i < n) parts[i] = fdec(penc[i]);
}

// ---------------- final sigmoid ---------------------------------------------

__global__ void k_sigmoid(const float* __restrict__ h, float* __restrict__ out, long n) {
  long i = (long)blockIdx.x * blockDim.x + threadIdx.x;
  if (i < n) out[i] = 1.f / (1.f + __expf(-h[i]));
}

// ---------------------------------------------------------------------------

static inline int cdiv_l(long a, long b) { return (int)((a + b - 1) / b); }

extern "C" void kernel_launch(void* const* d_in, const int* in_sizes, int n_in,
                              void* d_out, int out_size, void* d_ws, size_t ws_size,
                              hipStream_t stream) {
  (void)in_sizes; (void)n_in; (void)out_size; (void)ws_size;

  const float* x         = (const float*)d_in[0];      // [NN, F_IN]
  const int*   ei        = (const int*)  d_in[1];      // [2, NE]
  const int*   ga        = (const int*)  d_in[2];      // [NN]
  const float* glob_init = (const float*)d_in[3];      // [1, GDIM]

  const int dins [3] = {F_IN, DCONV, DCONV};
  const int douts[3] = {DCONV, DCONV, LDIM};

  // workspace carve-out
  char* ws = (char*)d_ws;
  size_t off = 0;
  auto carve = [&](size_t bytes) -> void* {
    void* p = ws + off;
    off = (off + bytes + 255) & ~(size_t)255;
    return p;
  };
  float*          dinv   = (float*)         carve((size_t)NN * 4);
  unsigned short* hb     = (unsigned short*)carve((size_t)NN * DCONV * 2);
  float*          h_lin  = (float*)         carve((size_t)NN * DCONV * 4);
  float*          h_agg  = (float*)         carve((size_t)NN * DCONV * 4);
  unsigned short* wbT    = (unsigned short*)carve((size_t)DCONV * DCONV * 2);
  float*          gnode  = (float*)         carve((size_t)NGRAPH * DCONV * 4);
  float*          glob_a = (float*)         carve((size_t)NGRAPH * GDIM * 4);
  float*          glob_b = (float*)         carve((size_t)NGRAPH * GDIM * 4);
  unsigned int*   penc   = (unsigned int*)  carve((size_t)NGRAPH * DCONV * 4);
  float*          parts  = (float*)         carve((size_t)NGRAPH * DCONV * 4);

  const int T = 256;

  // ---- preprocessing: degrees (with self loop), dinv, tiled global state
  k_fill_f32<<<cdiv_l(NN, T), T, 0, stream>>>(dinv, NN, 1.0f);
  k_deg<<<cdiv_l(NE, T), T, 0, stream>>>(ei, dinv);
  k_rsqrt_inplace<<<cdiv_l(NN, T), T, 0, stream>>>(dinv, NN);
  k_tile_glob<<<cdiv_l((long)NGRAPH * GDIM, T), T, 0, stream>>>(glob_init, glob_a);

  // layer-0 A operand: bf16(x)
  k_f32_to_bf16<<<cdiv_l((long)NN * F_IN, T), T, 0, stream>>>(x, hb, (long)NN * F_IN, 0);

  for (int li = 0; li < 3; ++li) {
    const int din = dins[li], dout = douts[li];
    const float* Wnn = (const float*)d_in[4 + li * 8 + 0];
    const float* bnn = (const float*)d_in[4 + li * 8 + 1];
    const float* Wgn = (const float*)d_in[4 + li * 8 + 2];
    const float* bgn = (const float*)d_in[4 + li * 8 + 3];
    const float* Wgg = (const float*)d_in[4 + li * 8 + 4];
    const float* bgg = (const float*)d_in[4 + li * 8 + 5];
    const float* Wng = (const float*)d_in[4 + li * 8 + 6];
    const float* bng = (const float*)d_in[4 + li * 8 + 7];

    // weights -> bf16, transposed [dout, din]
    k_wT_bf16<<<cdiv_l((long)din * dout, T), T, 0, stream>>>(Wnn, wbT, din, dout);
    // gnode = glob_a @ Wgn + bgn                         [NGRAPH, dout]
    k_small_gemm<<<cdiv_l((long)NGRAPH * dout, T), T, 0, stream>>>(
        glob_a, Wgn, bgn, gnode, NGRAPH, GDIM, dout, 0);

    // h_lin = hb @ Wnn + bnn + gnode[ga]                 [NN, dout]  (WMMA)
    {
      int mBlocks = (NN + 127) / 128;
      int nBlocks = dout / 64;
      k_wmma_gemm<<<mBlocks * nBlocks, 256, 0, stream>>>(
          hb, wbT, bnn, gnode, ga, h_lin, NN, din, dout);
    }

    // h_agg = scatter_add(norm * h_lin[row] -> col)
    k_fill_f32<<<cdiv_l((long)NN * dout, T), T, 0, stream>>>(h_agg, (long)NN * dout, 0.f);
    {
      long total = (long)(NE + NN) * (dout / 4);
      k_scatter<<<cdiv_l(total, T), T, 0, stream>>>(ei, dinv, h_lin, h_agg, dout, total);
    }

    // parts = segment_max(h_agg, ga)
    k_fill_u32<<<cdiv_l((long)NGRAPH * dout, T), T, 0, stream>>>(penc, (long)NGRAPH * dout, 0u);
    k_segmax<<<cdiv_l((long)NN * dout, T), T, 0, stream>>>(h_agg, ga, penc, dout, (long)NN * dout);
    k_decode<<<cdiv_l((long)NGRAPH * dout, T), T, 0, stream>>>(penc, parts, (long)NGRAPH * dout);

    // glob update: glob_b = glob_a @ Wgg + bgg + parts @ Wng + bng
    k_small_gemm<<<cdiv_l((long)NGRAPH * GDIM, T), T, 0, stream>>>(
        glob_a, Wgg, bgg, glob_b, NGRAPH, GDIM, GDIM, 0);
    k_small_gemm<<<cdiv_l((long)NGRAPH * GDIM, T), T, 0, stream>>>(
        parts, Wng, bng, glob_b, NGRAPH, dout, GDIM, 1);
    { float* t = glob_a; glob_a = glob_b; glob_b = t; }

    if (li < 2) {
      // next layer A operand: bf16(relu(h_agg))
      k_f32_to_bf16<<<cdiv_l((long)NN * dout, T), T, 0, stream>>>(
          h_agg, hb, (long)NN * dout, 1);
    } else {
      k_sigmoid<<<cdiv_l((long)NN * LDIM, T), T, 0, stream>>>(
          h_agg, (float*)d_out, (long)NN * LDIM);
    }
  }
}
